// SelfAttention_79843442032780
// MI455X (gfx1250) — matrix operand order repaired and checked by reading, not compiled, not run on hardware
//
#include <hip/hip_runtime.h>
#include <hip/hip_bf16.h>
#include <math.h>

typedef __attribute__((ext_vector_type(16))) __bf16 v16bf;
typedef __attribute__((ext_vector_type(8)))  float  v8f;

#define DIM  2048
#define NH   16
#define NKV  8
#define HD   128
#define NB   2
#define LSEQ 2048
#define BL   (NB * LSEQ) /* 4096 rows */

// ---------------- bf16 helpers (bit-exact, no __bf16 arithmetic) -------------
__device__ __forceinline__ __bf16 bf_from_bits(unsigned short u) {
  __bf16 r; __builtin_memcpy(&r, &u, 2); return r;
}
__device__ __forceinline__ float bf2f(__bf16 b) {
  unsigned short u; __builtin_memcpy(&u, &b, 2);
  unsigned x = ((unsigned)u) << 16;
  return __builtin_bit_cast(float, x);
}
__device__ __forceinline__ __bf16 f2bf(float f) {
  unsigned u = __builtin_bit_cast(unsigned, f);
  u += 0x7fffu + ((u >> 16) & 1u); // round-to-nearest-even
  return bf_from_bits((unsigned short)(u >> 16));
}

// A-fragment K index for 16-bit 16x32 A: contiguous, k = 16*g + 2*vv (matches B)
__device__ __forceinline__ int a_kbase(int vv, int g) {
  return 2 * vv + (g ? 16 : 0);
}

// ---------------- fp32 -> bf16 convert --------------------------------------
__global__ void f32_to_bf16(const float* __restrict__ in, __bf16* __restrict__ out, int n) {
  int i = blockIdx.x * blockDim.x + threadIdx.x;
  if (i < n) out[i] = f2bf(in[i]);
}

// ---------------- RoPE (in place, bf16), folds optional score scale ---------
__global__ void rope_scale(__bf16* X, const float* __restrict__ freqs,
                           int nheads, float scale, int total) {
  int idx = blockIdx.x * blockDim.x + threadIdx.x;
  if (idx >= total) return;
  int p  = idx & 63;          // pair index within head (HD/2 = 64)
  int t  = idx >> 6;
  int hh = t % nheads;
  int bl = t / nheads;        // b*L + l
  int l  = bl & (LSEQ - 1);
  size_t base = ((size_t)bl * nheads + hh) * HD + p * 2;
  float x0 = bf2f(X[base]), x1 = bf2f(X[base + 1]);
  float c  = freqs[((size_t)l * 64 + p) * 2 + 0];
  float sn = freqs[((size_t)l * 64 + p) * 2 + 1];
  X[base]     = f2bf((x0 * c - x1 * sn) * scale);
  X[base + 1] = f2bf((x0 * sn + x1 * c) * scale);
}

// ---------------- tiled bf16 WMMA GEMM: C(MxN) = A(MxK) * B(KxN) ------------
// block: 256 threads (8 waves); block tile 128x128; K-step 32.
// Each wave owns a 16x128 strip: 8 WMMA accumulators, A-fragment reused 8x.
template <bool BF16OUT>
__global__ __launch_bounds__(256) void gemm_bf16(const __bf16* __restrict__ A,
                                                 const __bf16* __restrict__ Bw,
                                                 void* __restrict__ Cout,
                                                 int M, int N, int Kd) {
  __shared__ __bf16 ldsA[128 * 40];   // 128 rows x 32 cols, stride 40 (16B rows)
  __shared__ __bf16 ldsBt[128 * 36];  // B transposed: 128 n-rows x 32 k-cols
  const int tid  = threadIdx.x;
  const int wave = tid >> 5, lane = tid & 31;
  const int g = lane >> 4, ln = lane & 15;
  const int n0 = blockIdx.x * 128, m0 = blockIdx.y * 128;

  v8f zero = {};
  v8f acc[8];
  for (int t = 0; t < 8; ++t) acc[t] = zero;

  for (int k0 = 0; k0 < Kd; k0 += 32) {
    __syncthreads();
    // stage A tile 128x32 (512 uint4 chunks, 2 per thread)
    #pragma unroll
    for (int i = 0; i < 2; ++i) {
      int chunk = tid + i * 256;
      int row = chunk >> 2, c8 = chunk & 3;
      uint4 d = *(const uint4*)(A + (size_t)(m0 + row) * Kd + k0 + c8 * 8);
      *(uint4*)&ldsA[row * 40 + c8 * 8] = d;
    }
    // stage B tile 32x128 transposed (512 uint4 chunks, u16 scatter)
    #pragma unroll
    for (int i = 0; i < 2; ++i) {
      int chunk = tid + i * 256;
      int row = chunk >> 4, c8 = chunk & 15;   // 16 chunks per k-row
      uint4 d = *(const uint4*)(Bw + (size_t)(k0 + row) * N + n0 + c8 * 8);
      unsigned w[4] = {d.x, d.y, d.z, d.w};
      #pragma unroll
      for (int j = 0; j < 8; ++j) {
        unsigned short hb = (unsigned short)((w[j >> 1] >> ((j & 1) * 16)) & 0xffffu);
        ldsBt[(c8 * 8 + j) * 36 + row] = bf_from_bits(hb);
      }
    }
    // prefetch next K-step tiles into cache (global_prefetch_b8)
    if (k0 + 32 < Kd) {
      #pragma unroll
      for (int i = 0; i < 2; ++i) {
        int chunk = tid + i * 256;
        __builtin_prefetch(A + (size_t)(m0 + (chunk >> 2)) * Kd + (k0 + 32) + (chunk & 3) * 8, 0, 1);
        __builtin_prefetch(Bw + (size_t)(k0 + 32 + (chunk >> 4)) * N + n0 + (chunk & 15) * 8, 0, 1);
      }
    }
    __syncthreads();

    union { v16bf v; unsigned u[8]; } af;
    #pragma unroll
    for (int vv = 0; vv < 8; ++vv)
      af.u[vv] = *(const unsigned*)&ldsA[(wave * 16 + ln) * 40 + a_kbase(vv, g)];
    #pragma unroll
    for (int t = 0; t < 8; ++t) {
      union { v16bf v; unsigned u[8]; } bfr;
      #pragma unroll
      for (int p = 0; p < 8; ++p)
        bfr.u[p] = *(const unsigned*)&ldsBt[(t * 16 + ln) * 36 + 2 * p + (g ? 16 : 0)];
      acc[t] = __builtin_amdgcn_wmma_f32_16x16x32_bf16(
          false, af.v, false, bfr.v, (short)0, acc[t], false, false);
    }
  }

  #pragma unroll
  for (int t = 0; t < 8; ++t)
    #pragma unroll
    for (int r = 0; r < 8; ++r) {
      size_t idx = (size_t)(m0 + wave * 16 + r + 8 * g) * N + n0 + t * 16 + ln;
      if (BF16OUT) ((__bf16*)Cout)[idx] = f2bf(acc[t][r]);
      else         ((float*)Cout)[idx]  = acc[t][r];
    }
}

// ---------------- flash attention: per (128 q-rows, b*h) block --------------
// Q pre-scaled by 1/sqrt(HD). Layouts: Q (B,L,NH,HD), K/V (B,L,NKV,HD), bf16.
__global__ __launch_bounds__(256) void attn(const __bf16* __restrict__ Qb,
                                            const __bf16* __restrict__ Kb,
                                            const __bf16* __restrict__ Vb,
                                            __bf16* __restrict__ Ob) {
  __shared__ __bf16 ldsK[32 * 132];    // 32 keys x 128 dims (stride 132)
  __shared__ __bf16 ldsVt[128 * 36];   // V transposed: 128 dims x 32 keys (stride 36)
  __shared__ __bf16 ldsP[8][16 * 34];  // per-wave P tile 16 x 32 (stride 34)

  const int qt = blockIdx.x;           // 0..15
  const int bh = blockIdx.y;           // 0..31
  const int b = bh / NH, h = bh % NH;
  const int kvh = h >> 1;              // N_REP = 2
  const int tid  = threadIdx.x;
  const int wave = tid >> 5, lane = tid & 31;
  const int g = lane >> 4, ln = lane & 15;
  const int q_base = qt * 128 + wave * 16;

  // preload this wave's 16x128 Q strip as 4 A-fragments
  v16bf qf[4];
  {
    const __bf16* qrow = Qb + (((size_t)b * LSEQ + (q_base + ln)) * NH + h) * HD;
    #pragma unroll
    for (int dc = 0; dc < 4; ++dc) {
      union { v16bf v; unsigned u[8]; } tq;
      #pragma unroll
      for (int vv = 0; vv < 8; ++vv)
        tq.u[vv] = *(const unsigned*)(qrow + dc * 32 + a_kbase(vv, g));
      qf[dc] = tq.v;
    }
  }

  v8f zero = {};
  v8f acc[8];
  for (int t = 0; t < 8; ++t) acc[t] = zero;
  float m_i[8], l_i[8];
  for (int r = 0; r < 8; ++r) { m_i[r] = -__builtin_inff(); l_i[r] = 0.0f; }

  const int kend = qt * 128 + 128;     // causal horizon for this q tile
  for (int kc = 0; kc < kend; kc += 32) {
    __syncthreads();
    // stage K chunk (32 keys x 128 dims)
    #pragma unroll
    for (int i = 0; i < 2; ++i) {
      int chunk = tid + i * 256;
      int key = chunk >> 4, c8 = chunk & 15;
      const __bf16* src = Kb + (((size_t)b * LSEQ + (kc + key)) * NKV + kvh) * HD + c8 * 8;
      uint4 d = *(const uint4*)src;
      unsigned* dst = (unsigned*)&ldsK[key * 132 + c8 * 8];
      dst[0] = d.x; dst[1] = d.y; dst[2] = d.z; dst[3] = d.w;
    }
    // stage V chunk transposed (dims x keys)
    #pragma unroll
    for (int i = 0; i < 2; ++i) {
      int chunk = tid + i * 256;
      int key = chunk >> 4, c8 = chunk & 15;
      const __bf16* src = Vb + (((size_t)b * LSEQ + (kc + key)) * NKV + kvh) * HD + c8 * 8;
      uint4 d = *(const uint4*)src;
      unsigned w[4] = {d.x, d.y, d.z, d.w};
      #pragma unroll
      for (int j = 0; j < 8; ++j) {
        unsigned short hb = (unsigned short)((w[j >> 1] >> ((j & 1) * 16)) & 0xffffu);
        ldsVt[(c8 * 8 + j) * 36 + key] = bf_from_bits(hb);
      }
    }
    // prefetch next chunk of K/V (global_prefetch_b8)
    if (kc + 32 < kend) {
      #pragma unroll
      for (int i = 0; i < 2; ++i) {
        int chunk = tid + i * 256;
        int key = chunk >> 4, c8 = chunk & 15;
        size_t nbase = (((size_t)b * LSEQ + (kc + 32 + key)) * NKV + kvh) * HD + c8 * 8;
        __builtin_prefetch(Kb + nbase, 0, 1);
        __builtin_prefetch(Vb + nbase, 0, 1);
      }
    }
    __syncthreads();

    // S = Q * K^T for two 16-key tiles (4 d-chunks each)
    v8f s[2]; s[0] = zero; s[1] = zero;
    #pragma unroll
    for (int dc = 0; dc < 4; ++dc)
      #pragma unroll
      for (int t2 = 0; t2 < 2; ++t2) {
        union { v16bf v; unsigned u[8]; } bb;
        #pragma unroll
        for (int p = 0; p < 8; ++p)
          bb.u[p] = *(const unsigned*)&ldsK[(t2 * 16 + ln) * 132 + dc * 32 + 2 * p + (g ? 16 : 0)];
        s[t2] = __builtin_amdgcn_wmma_f32_16x16x32_bf16(
            false, qf[dc], false, bb.v, (short)0, s[t2], false, false);
      }

    // causal mask
    #pragma unroll
    for (int t2 = 0; t2 < 2; ++t2) {
      int key = kc + t2 * 16 + ln;
      #pragma unroll
      for (int r = 0; r < 8; ++r)
        if (key > q_base + r + 8 * g) s[t2][r] = -__builtin_inff();
    }
    // online softmax (row reductions across the 16-lane half-groups)
    #pragma unroll
    for (int r = 0; r < 8; ++r) {
      float mx = fmaxf(s[0][r], s[1][r]);
      mx = fmaxf(mx, __shfl_xor(mx, 1, 16));
      mx = fmaxf(mx, __shfl_xor(mx, 2, 16));
      mx = fmaxf(mx, __shfl_xor(mx, 4, 16));
      mx = fmaxf(mx, __shfl_xor(mx, 8, 16));
      float m_new = fmaxf(m_i[r], mx);
      float corr  = __expf(m_i[r] - m_new);
      m_i[r] = m_new;
      float p0 = __expf(s[0][r] - m_new);
      float p1 = __expf(s[1][r] - m_new);
      s[0][r] = p0; s[1][r] = p1;
      l_i[r] = l_i[r] * corr + p0 + p1;
      #pragma unroll
      for (int t = 0; t < 8; ++t) acc[t][r] *= corr;
    }

    // P: C-layout -> A-layout via wave-private LDS round trip
    #pragma unroll
    for (int t2 = 0; t2 < 2; ++t2)
      #pragma unroll
      for (int r = 0; r < 8; ++r)
        ldsP[wave][(r + 8 * g) * 34 + t2 * 16 + ln] = f2bf(s[t2][r]);
    __builtin_amdgcn_wave_barrier();  // same-wave DS ops are in-order; block reordering
    union { v16bf v; unsigned u[8]; } pa;
    #pragma unroll
    for (int vv = 0; vv < 8; ++vv)
      pa.u[vv] = *(const unsigned*)&ldsP[wave][ln * 34 + a_kbase(vv, g)];

    // acc += P(16x32) * V(32x128): 8 n-tiles
    #pragma unroll
    for (int t = 0; t < 8; ++t) {
      union { v16bf v; unsigned u[8]; } vb;
      #pragma unroll
      for (int p = 0; p < 8; ++p)
        vb.u[p] = *(const unsigned*)&ldsVt[(t * 16 + ln) * 36 + 2 * p + (g ? 16 : 0)];
      acc[t] = __builtin_amdgcn_wmma_f32_16x16x32_bf16(
          false, pa.v, false, vb.v, (short)0, acc[t], false, false);
    }
  }

  // finalize: row-sum reduce, normalize, store (B,L,NH,HD) bf16
  #pragma unroll
  for (int r = 0; r < 8; ++r) {
    float l = l_i[r];
    l += __shfl_xor(l, 1, 16);
    l += __shfl_xor(l, 2, 16);
    l += __shfl_xor(l, 4, 16);
    l += __shfl_xor(l, 8, 16);
    l_i[r] = 1.0f / l;
  }
  #pragma unroll
  for (int t = 0; t < 8; ++t)
    #pragma unroll
    for (int r = 0; r < 8; ++r) {
      int q = q_base + r + 8 * g;
      Ob[(((size_t)b * LSEQ + q) * NH + h) * HD + t * 16 + ln] = f2bf(acc[t][r] * l_i[r]);
    }
}

// ---------------- host-side orchestration -----------------------------------
extern "C" void kernel_launch(void* const* d_in, const int* in_sizes, int n_in,
                              void* d_out, int out_size, void* d_ws, size_t ws_size,
                              hipStream_t stream) {
  (void)in_sizes; (void)n_in; (void)out_size; (void)ws_size;
  const float* x  = (const float*)d_in[0];
  const float* fr = (const float*)d_in[1];  // (L, 64, 2)
  const float* wq = (const float*)d_in[2];
  const float* wk = (const float*)d_in[3];
  const float* wv = (const float*)d_in[4];
  const float* wo = (const float*)d_in[5];

  char* ws = (char*)d_ws;
  size_t off = 0;
  auto alloc = [&](size_t bytes) { size_t o = off; off += (bytes + 255) & ~(size_t)255; return o; };
  __bf16* x_bf  = (__bf16*)(ws + alloc((size_t)BL * DIM * 2));
  __bf16* wq_bf = (__bf16*)(ws + alloc((size_t)DIM * DIM * 2));
  __bf16* wk_bf = (__bf16*)(ws + alloc((size_t)DIM * NKV * HD * 2));
  __bf16* wv_bf = (__bf16*)(ws + alloc((size_t)DIM * NKV * HD * 2));
  __bf16* wo_bf = (__bf16*)(ws + alloc((size_t)DIM * DIM * 2));
  __bf16* q_bf  = (__bf16*)(ws + alloc((size_t)BL * DIM * 2));
  __bf16* k_bf  = (__bf16*)(ws + alloc((size_t)BL * NKV * HD * 2));
  __bf16* v_bf  = (__bf16*)(ws + alloc((size_t)BL * NKV * HD * 2));
  __bf16* ao_bf = (__bf16*)(ws + alloc((size_t)BL * DIM * 2));

  auto cvt = [&](const float* in, __bf16* outp, int n) {
    f32_to_bf16<<<(n + 255) / 256, 256, 0, stream>>>(in, outp, n);
  };
  cvt(x,  x_bf,  BL * DIM);
  cvt(wq, wq_bf, DIM * DIM);
  cvt(wk, wk_bf, DIM * NKV * HD);
  cvt(wv, wv_bf, DIM * NKV * HD);
  cvt(wo, wo_bf, DIM * DIM);

  // projections (bf16 out); block tile 128x128
  gemm_bf16<true><<<dim3(DIM / 128, BL / 128), 256, 0, stream>>>(x_bf, wq_bf, q_bf, BL, DIM, DIM);
  gemm_bf16<true><<<dim3((NKV * HD) / 128, BL / 128), 256, 0, stream>>>(x_bf, wk_bf, k_bf, BL, NKV * HD, DIM);
  gemm_bf16<true><<<dim3((NKV * HD) / 128, BL / 128), 256, 0, stream>>>(x_bf, wv_bf, v_bf, BL, NKV * HD, DIM);

  // RoPE (Q also absorbs 1/sqrt(HD) score scale)
  const float qscale = 0.08838834764831845f; // 1/sqrt(128)
  int qtot = NB * LSEQ * NH * (HD / 2);
  rope_scale<<<(qtot + 255) / 256, 256, 0, stream>>>(q_bf, fr, NH, qscale, qtot);
  int ktot = NB * LSEQ * NKV * (HD / 2);
  rope_scale<<<(ktot + 255) / 256, 256, 0, stream>>>(k_bf, fr, NKV, 1.0f, ktot);

  // attention
  attn<<<dim3(LSEQ / 128, NB * NH), 256, 0, stream>>>(q_bf, k_bf, v_bf, ao_bf);

  // output projection (fp32 out)
  gemm_bf16<false><<<dim3(DIM / 128, BL / 128), 256, 0, stream>>>(ao_bf, wo_bf, (float*)d_out, BL, DIM, DIM);
}
